// FCTEncoderLayer_30511447671479
// MI455X (gfx1250) — matrix operand order repaired and checked by reading, not compiled, not run on hardware
//
#include <hip/hip_runtime.h>
#include <hip/hip_bf16.h>

// ---------------------------------------------------------------------------
// FCT encoder layer for MI455X (gfx1250): bf16 WMMA GEMMs + flash attention
// Round 5: fix cvt_f32_ubyte builtin (use uitofp pattern the backend matches
// to v_cvt_f32_ubyteN). exp2-domain softmax, async-LDS K/V double buffering.
// ---------------------------------------------------------------------------

typedef __bf16 bf16_t;
typedef __attribute__((ext_vector_type(16))) __bf16 v16bf;
typedef __attribute__((ext_vector_type(8)))  float  v8f;

#define D_MODEL  1024
#define DFF      4096
#define NHEAD    16
#define DHEAD    64
#define LQ       1024
#define BATCH    2
#define NLAYER   5                      // 4 bank slabs + current
#define LK_TOT   (NLAYER * LQ)          // 5120
#define ROWS_Q   (BATCH * LQ)           // 2048
#define ROWS_MEM (NLAYER * BATCH * LQ)  // 10240

// ---------------------------------------------------------------------------
// WMMA helpers (CDNA5 layouts, ISA 7.12.2)
// ---------------------------------------------------------------------------

__device__ __forceinline__ v8f wmma_bf16(v16bf a, v16bf b, v8f c) {
  // 8-arg form: (neg_a, A, neg_b, B, c_mod, C, reuse_a, reuse_b)
  return __builtin_amdgcn_wmma_f32_16x16x32_bf16(false, a, false, b,
                                                 (short)0, c, false, false);
}

union BF16x16 {
  v16bf v;
  uint4 q[2];
};

// Contiguous 32-byte per-lane fragment load (two global/ds b128 ops).
__device__ __forceinline__ v16bf load_frag_contig(const bf16_t* p) {
  BF16x16 u;
  u.q[0] = *(const uint4*)p;
  u.q[1] = *(const uint4*)(p + 8);
  return u.v;
}

// A fragment, 16x32 bf16, row-major source with leading dim `ld`.
__device__ __forceinline__ v16bf load_a_frag(const bf16_t* p, int ld) {
  const int lane = threadIdx.x & 31;
  const int row  = lane & 15;
  const int half = lane >> 4;
  v16bf a;
#pragma unroll
  for (int e = 0; e < 16; ++e) {
    const int k = half * 16 + e;
    a[e] = p[row * ld + k];
  }
  return a;
}

// B fragment, 32x16 bf16, row-major [K][N] source with leading dim `ld`.
__device__ __forceinline__ v16bf load_b_frag(const bf16_t* p, int ld) {
  const int lane = threadIdx.x & 31;
  const int n    = lane & 15;
  const int kb   = (lane >> 4) * 16;
  v16bf b;
#pragma unroll
  for (int e = 0; e < 16; ++e) b[e] = p[(kb + e) * ld + n];
  return b;
}

// B fragment for Q*K^T from K rows [kpos][dh] (LDS tile):
// B[kk][n] = K[row0+n][koff+kk] -> per lane contiguous 32B.
__device__ __forceinline__ v16bf load_b_kT(const bf16_t* kv, int koff) {
  const int lane = threadIdx.x & 31;
  const int n    = lane & 15;
  const int kb   = (lane >> 4) * 16;
  return load_frag_contig(kv + n * DHEAD + koff + kb);
}

// B fragment for P*V from LDS-staged transposed V tile [dh=64][kk=32]:
// B[kk][n] = Vt[col0+n][kk] -> per lane contiguous 32B.
__device__ __forceinline__ v16bf load_b_vT_lds(const bf16_t* sv, int col0) {
  const int lane = threadIdx.x & 31;
  const int n    = lane & 15;
  const int kb   = (lane >> 4) * 16;
  return load_frag_contig(sv + (col0 + n) * 32 + kb);
}

// byte N of word -> float; backend pattern-matches this to v_cvt_f32_ubyteN
__device__ __forceinline__ float cvt_ubyte(unsigned int w, int b) {
  return (float)((w >> (8 * b)) & 0xffu);
}

// ---------------------------------------------------------------------------
// CDNA5 async global->LDS copy (tracked by ASYNCcnt).
// lds_off = wave-relative LDS byte address (low 32 bits of generic pointer).
// ---------------------------------------------------------------------------
__device__ __forceinline__ void async_load_b128(unsigned int lds_off,
                                                const bf16_t* g) {
  asm volatile("global_load_async_to_lds_b128 %0, %1, off"
               :: "v"(lds_off), "v"(g)
               : "memory");
}

// ---------------------------------------------------------------------------
// Elementwise f32 -> bf16 convert
// ---------------------------------------------------------------------------
__global__ __launch_bounds__(256) void f32_to_bf16_kernel(
    const float* __restrict__ in, bf16_t* __restrict__ out, int n) {
  int i = blockIdx.x * 256 + threadIdx.x;
  if (i < n) out[i] = (bf16_t)in[i];
}

// ---------------------------------------------------------------------------
// Mask transpose: [b][q][k] bytes -> [b][k][q] bytes
// ---------------------------------------------------------------------------
__global__ __launch_bounds__(256) void mask_transpose_kernel(
    const unsigned char* __restrict__ in, unsigned char* __restrict__ out) {
  const int i = blockIdx.x * 256 + threadIdx.x;  // over BATCH*LQ*LQ
  const int b = i >> 20;
  const int q = (i >> 10) & (LQ - 1);
  const int k = i & (LQ - 1);
  out[(((size_t)b << 10) | k) << 10 | q] = in[i];
}

// ---------------------------------------------------------------------------
// LayerNorm over rows of 1024; one block (8 waves) per row.
// ---------------------------------------------------------------------------
__global__ __launch_bounds__(256) void ln_kernel(
    const float* __restrict__ x, const float* __restrict__ g,
    const float* __restrict__ bp, float* __restrict__ out_f32,
    bf16_t* __restrict__ out_b16) {
  const int row = blockIdx.x;
  const float* xr = x + (size_t)row * D_MODEL;
  float v4[4];
  float s = 0.f;
#pragma unroll
  for (int i = 0; i < 4; ++i) { v4[i] = xr[threadIdx.x + i * 256]; s += v4[i]; }
#pragma unroll
  for (int d = 1; d < 32; d <<= 1) s += __shfl_xor(s, d, 32);
  __shared__ float red1[8];
  __shared__ float red2[8];
  const int wid = threadIdx.x >> 5, lane = threadIdx.x & 31;
  if (lane == 0) red1[wid] = s;
  __syncthreads();
  float mean = 0.f;
#pragma unroll
  for (int i = 0; i < 8; ++i) mean += red1[i];
  mean *= (1.0f / D_MODEL);
  float vs = 0.f;
#pragma unroll
  for (int i = 0; i < 4; ++i) { float d0 = v4[i] - mean; vs += d0 * d0; }
#pragma unroll
  for (int d = 1; d < 32; d <<= 1) vs += __shfl_xor(vs, d, 32);
  if (lane == 0) red2[wid] = vs;
  __syncthreads();
  float var = 0.f;
#pragma unroll
  for (int i = 0; i < 8; ++i) var += red2[i];
  const float rstd = rsqrtf(var * (1.0f / D_MODEL) + 1e-5f);
#pragma unroll
  for (int i = 0; i < 4; ++i) {
    const int c = threadIdx.x + i * 256;
    const float o = (v4[i] - mean) * rstd * g[c] + bp[c];
    if (out_f32) out_f32[(size_t)row * D_MODEL + c] = o;
    if (out_b16) out_b16[(size_t)row * D_MODEL + c] = (bf16_t)o;
  }
}

// ---------------------------------------------------------------------------
// Generic bf16 WMMA GEMM:  C[M,N] = A[M,K] @ B[K,N]  (row-major bf16)
// Block tile 64x128, 8 waves, each wave 32x32 (4 accumulators), BK=32.
// Epilogues: 0=pq  1=k-permute  2=residual->f32  3=bias+relu->bf16
//            4=bias+residual->f32  5=v transposed
// ---------------------------------------------------------------------------
template <int EPI>
__global__ __launch_bounds__(256) void gemm_bf16_wmma(
    const bf16_t* __restrict__ A, const bf16_t* __restrict__ B,
    const float* __restrict__ bias, const float* __restrict__ residual,
    float scale, int M, int N, int K,
    float* __restrict__ outf, bf16_t* __restrict__ outb) {
  __shared__ alignas(16) bf16_t sA[64 * 32];
  __shared__ alignas(16) bf16_t sB[32 * 128];

  const int wid  = threadIdx.x >> 5;
  const int lane = threadIdx.x & 31;
  const int wrow = wid >> 2;
  const int wcol = wid & 3;
  const int blockM = blockIdx.y * 64;
  const int blockN = blockIdx.x * 128;

  v8f acc[2][2] = {{{0.f}, {0.f}}, {{0.f}, {0.f}}};

  for (int k0 = 0; k0 < K; k0 += 32) {
    {  // stage A tile 64x32 (8 bf16 per thread)
      const int r = threadIdx.x >> 2, c = (threadIdx.x & 3) * 8;
      const bf16_t* src = A + (size_t)(blockM + r) * K + k0 + c;
      *(uint4*)&sA[r * 32 + c] = *(const uint4*)src;
    }
    {  // stage B tile 32x128 (16 bf16 per thread)
      const int r = threadIdx.x >> 3, c = (threadIdx.x & 7) * 16;
      const bf16_t* src = B + (size_t)(k0 + r) * N + blockN + c;
      *(uint4*)&sB[r * 128 + c]     = *(const uint4*)src;
      *(uint4*)&sB[r * 128 + c + 8] = *(const uint4*)(src + 8);
    }
    __syncthreads();
    v16bf af0 = load_a_frag(&sA[(wrow * 32 + 0)  * 32], 32);
    v16bf af1 = load_a_frag(&sA[(wrow * 32 + 16) * 32], 32);
    v16bf bf0 = load_b_frag(&sB[wcol * 32 + 0],  128);
    v16bf bf1 = load_b_frag(&sB[wcol * 32 + 16], 128);
    acc[0][0] = wmma_bf16(af0, bf0, acc[0][0]);
    acc[0][1] = wmma_bf16(af0, bf1, acc[0][1]);
    acc[1][0] = wmma_bf16(af1, bf0, acc[1][0]);
    acc[1][1] = wmma_bf16(af1, bf1, acc[1][1]);
    __syncthreads();
  }

  const int half = lane >> 4, n = lane & 15;
  const int rbase = blockM + wrow * 32;
  const int cbase = blockN + wcol * 32;
#pragma unroll
  for (int i = 0; i < 2; ++i)
#pragma unroll
    for (int j = 0; j < 2; ++j)
#pragma unroll
      for (int v = 0; v < 8; ++v) {
        const int row = rbase + i * 16 + v + 8 * half;
        const int col = cbase + j * 16 + n;
        float val = acc[i][j][v] * scale;
        if constexpr (EPI == 0) {  // pq: [b,h,lq,dh]
          const int b = row >> 10, q = row & (LQ - 1);
          const int h = col >> 6, dd = col & (DHEAD - 1);
          outb[(((size_t)(b * NHEAD + h) * LQ + q) * DHEAD) + dd] = (bf16_t)val;
        } else if constexpr (EPI == 1) {  // k: [b,h,L*lk,dh]
          const int l = row >> 11, b = (row >> 10) & 1, lk = row & (LQ - 1);
          const int h = col >> 6, dd = col & (DHEAD - 1);
          const int kpos = l * LQ + lk;
          outb[(((size_t)(b * NHEAD + h) * LK_TOT + kpos) * DHEAD) + dd] = (bf16_t)val;
        } else if constexpr (EPI == 2) {  // attn_out + residual -> f32
          const size_t idx = (size_t)row * N + col;
          outf[idx] = residual[idx] + val;
        } else if constexpr (EPI == 3) {  // bias + relu -> bf16
          val += bias[col];
          outb[(size_t)row * N + col] = (bf16_t)fmaxf(val, 0.f);
        } else if constexpr (EPI == 4) {  // bias + residual -> f32 (final out)
          const size_t idx = (size_t)row * N + col;
          outf[idx] = residual[idx] + val + bias[col];
        } else {  // EPI == 5: v transposed [b,h,dh,L*lk]
          const int l = row >> 11, b = (row >> 10) & 1, lk = row & (LQ - 1);
          const int h = col >> 6, dd = col & (DHEAD - 1);
          const int kpos = l * LQ + lk;
          outb[(((size_t)(b * NHEAD + h) * DHEAD + dd) * LK_TOT) + kpos] = (bf16_t)val;
        }
      }
}

// ---------------------------------------------------------------------------
// Flash attention: one block = one (b, h) x 128 q rows; 8 waves each own a
// 16-row q tile and SHARE the K/V stream, staged into double-buffered LDS by
// async global->LDS b128 loads (one per thread per tile; ASYNCcnt pipeline).
// Softmax runs in exp2 domain (log2e pre-folded into the q projection scale).
// ---------------------------------------------------------------------------
__global__ __launch_bounds__(256) void attn_kernel(
    const bf16_t* __restrict__ pq, const bf16_t* __restrict__ pk,
    const bf16_t* __restrict__ pvt, const unsigned char* __restrict__ maskT,
    bf16_t* __restrict__ ctx) {
  __shared__ alignas(16) bf16_t sK[2][32 * 64];  // K tile: [kk 0..31][dh 0..63]
  __shared__ alignas(16) bf16_t sV[2][64 * 32];  // Vt tile: [dh 0..63][kk 0..31]
  __shared__ alignas(16) bf16_t sP[8][16 * 32];  // per-wave P tile

  const int wid  = threadIdx.x >> 5;
  const int lane = threadIdx.x & 31;
  const int b  = blockIdx.x >> 7;              // 256 blocks: [b][h][qchunk]
  const int h  = (blockIdx.x >> 3) & (NHEAD - 1);
  const int q0 = (blockIdx.x & 7) * 128 + wid * 16;
  const int half = lane >> 4, n = lane & 15;

  const size_t bh = (size_t)(b * NHEAD + h);
  const bf16_t* pqb  = pq  + (bh * LQ + q0) * DHEAD;
  const bf16_t* pkb  = pk  + bh * LK_TOT * DHEAD;
  const bf16_t* pvtb = pvt + bh * DHEAD * LK_TOT;
  const unsigned char* mtb = maskT + (size_t)b * LQ * LQ;

  const unsigned int sK_lds[2] = {(unsigned int)(size_t)&sK[0][0],
                                  (unsigned int)(size_t)&sK[1][0]};
  const unsigned int sV_lds[2] = {(unsigned int)(size_t)&sV[0][0],
                                  (unsigned int)(size_t)&sV[1][0]};

  // Async-stage K/V tiles for key block kq into buffer buf (1 b128/thread).
  auto issue_tiles = [&](int buf, int kq) {
    const bf16_t* kg = pkb + (size_t)(kq + (threadIdx.x >> 3)) * DHEAD +
                       (threadIdx.x & 7) * 8;
    async_load_b128(sK_lds[buf] + threadIdx.x * 16, kg);
    const bf16_t* vg = pvtb + (size_t)(threadIdx.x >> 2) * LK_TOT + kq +
                       (threadIdx.x & 3) * 8;
    async_load_b128(sV_lds[buf] + threadIdx.x * 16, vg);
  };

  // Q fragments (pq pre-scaled by log2e/sqrt(dh) -> scores in log2 domain)
  const v16bf aq0 = load_a_frag(pqb, DHEAD);
  const v16bf aq1 = load_a_frag(pqb + 32, DHEAD);

  float m[8], lsum[8];
#pragma unroll
  for (int v = 0; v < 8; ++v) { m[v] = -3.0e38f; lsum[v] = 0.f; }
  v8f oacc[4] = {{0.f}, {0.f}, {0.f}, {0.f}};

  issue_tiles(0, 0);  // prologue

  for (int k0 = 0; k0 < LK_TOT; k0 += 32) {
    const int cur = (k0 >> 5) & 1;
    const bool more = (k0 + 32) < LK_TOT;
    if (more) {
      issue_tiles(cur ^ 1, k0 + 32);
      asm volatile("s_wait_asynccnt 0x2" ::: "memory");  // cur's 2 loads done
    } else {
      asm volatile("s_wait_asynccnt 0x0" ::: "memory");
    }
    __syncthreads();  // all waves' staged data visible

    const bf16_t* sKc = &sK[cur][0];
    const bf16_t* sVc = &sV[cur][0];

    // -------- batched preloads: K frags, V frags, mask words --------
    v16bf bk[2][2], bv[4];
    unsigned int mrow[2][2];
#pragma unroll
    for (int j = 0; j < 2; ++j) {
      const bf16_t* kvp = sKc + (j * 16) * DHEAD;
      bk[j][0] = load_b_kT(kvp, 0);
      bk[j][1] = load_b_kT(kvp, 32);
      const int lk = (k0 + j * 16 + n) & (LQ - 1);
      const uint2 mb8 = *(const uint2*)(mtb + (size_t)lk * LQ + q0 + 8 * half);
      mrow[j][0] = mb8.x;
      mrow[j][1] = mb8.y;
    }
#pragma unroll
    for (int dcb = 0; dcb < 4; ++dcb) bv[dcb] = load_b_vT_lds(sVc, dcb * 16);

    // -------- scores: two 16x16 C tiles --------
    v8f s_t[2];
#pragma unroll
    for (int j = 0; j < 2; ++j)
      s_t[j] = wmma_bf16(aq1, bk[j][1], wmma_bf16(aq0, bk[j][0], v8f{0.f}));

    // -------- mask (cvt_ubyte + fma) + online exp2 softmax --------
#pragma unroll
    for (int v = 0; v < 8; ++v) {
      float s0 = s_t[0][v], s1 = s_t[1][v];
      s0 = fmaf(cvt_ubyte(mrow[0][v >> 2], v & 3), -3.0e38f, s0);
      s1 = fmaf(cvt_ubyte(mrow[1][v >> 2], v & 3), -3.0e38f, s1);
      float cm = fmaxf(s0, s1);
#pragma unroll
      for (int d = 1; d < 16; d <<= 1) cm = fmaxf(cm, __shfl_xor(cm, d, 32));
      const float mn = fmaxf(m[v], cm);
      const float alpha = __builtin_amdgcn_exp2f(m[v] - mn);
      const float p0 = __builtin_amdgcn_exp2f(s0 - mn);
      const float p1 = __builtin_amdgcn_exp2f(s1 - mn);
      float rs = p0 + p1;
#pragma unroll
      for (int d = 1; d < 16; d <<= 1) rs += __shfl_xor(rs, d, 32);
      lsum[v] = lsum[v] * alpha + rs;
      m[v] = mn;
#pragma unroll
      for (int dcb = 0; dcb < 4; ++dcb) oacc[dcb][v] *= alpha;
      const int r = v + 8 * half;
      sP[wid][r * 32 + n]      = (bf16_t)p0;
      sP[wid][r * 32 + 16 + n] = (bf16_t)p1;
    }
    // -------- ctx += P(16x32) x V(32x64) --------
    const v16bf ap = load_a_frag(&sP[wid][0], 32);
#pragma unroll
    for (int dcb = 0; dcb < 4; ++dcb)
      oacc[dcb] = wmma_bf16(ap, bv[dcb], oacc[dcb]);

    __syncthreads();  // buffer `cur` free for the k0+64 issue
  }

  // normalize and store ctx as [b, lq, h*64+dh] (GEMM-ready row-major)
#pragma unroll
  for (int v = 0; v < 8; ++v) {
    const int qrow = q0 + v + 8 * half;
    const float rl = 1.0f / lsum[v];
#pragma unroll
    for (int dcb = 0; dcb < 4; ++dcb) {
      const int dd = dcb * 16 + n;
      ctx[((size_t)(b * LQ + qrow)) * D_MODEL + h * DHEAD + dd] =
          (bf16_t)(oacc[dcb][v] * rl);
    }
  }
}

// ---------------------------------------------------------------------------
// Host orchestration
// ---------------------------------------------------------------------------
extern "C" void kernel_launch(void* const* d_in, const int* in_sizes, int n_in,
                              void* d_out, int out_size, void* d_ws,
                              size_t ws_size, hipStream_t stream) {
  (void)in_sizes; (void)n_in; (void)out_size; (void)ws_size;
  const float* x    = (const float*)d_in[0];
  const float* memb = (const float*)d_in[1];
  const unsigned char* mask = (const unsigned char*)d_in[2];
  const float* ln1g = (const float*)d_in[3];
  const float* ln1b = (const float*)d_in[4];
  const float* Wq   = (const float*)d_in[5];
  const float* Wk   = (const float*)d_in[6];
  const float* Wv   = (const float*)d_in[7];
  const float* Wo   = (const float*)d_in[8];
  const float* ln2g = (const float*)d_in[9];
  const float* ln2b = (const float*)d_in[10];
  const float* W1   = (const float*)d_in[11];
  const float* b1   = (const float*)d_in[12];
  const float* W2   = (const float*)d_in[13];
  const float* b2   = (const float*)d_in[14];

  float* outp    = (float*)d_out;
  float* mem_out = outp + (size_t)ROWS_Q * D_MODEL;              // [5,b,lq,d]
  float* q_f32   = mem_out + (size_t)4 * ROWS_Q * D_MODEL;       // slab 4 = q

  char* ws = (char*)d_ws;
  size_t off = 0;
  auto alloc = [&](size_t bytes) -> char* {
    char* p = ws + off;
    off += (bytes + 255) & ~(size_t)255;
    return p;
  };
  bf16_t* mem_b = (bf16_t*)alloc((size_t)ROWS_MEM * D_MODEL * 2);
  bf16_t* q_b   = mem_b + (size_t)4 * ROWS_Q * D_MODEL;          // slab 4
  bf16_t* Wq_b  = (bf16_t*)alloc((size_t)D_MODEL * D_MODEL * 2);
  bf16_t* Wk_b  = (bf16_t*)alloc((size_t)D_MODEL * D_MODEL * 2);
  bf16_t* Wv_b  = (bf16_t*)alloc((size_t)D_MODEL * D_MODEL * 2);
  bf16_t* Wo_b  = (bf16_t*)alloc((size_t)D_MODEL * D_MODEL * 2);
  bf16_t* W1_b  = (bf16_t*)alloc((size_t)D_MODEL * DFF * 2);
  bf16_t* W2_b  = (bf16_t*)alloc((size_t)DFF * D_MODEL * 2);
  bf16_t* pqw   = (bf16_t*)alloc((size_t)ROWS_Q * D_MODEL * 2);
  bf16_t* pkw   = (bf16_t*)alloc((size_t)BATCH * NHEAD * LK_TOT * DHEAD * 2);
  bf16_t* pvtw  = (bf16_t*)alloc((size_t)BATCH * NHEAD * LK_TOT * DHEAD * 2);
  bf16_t* ctxw  = (bf16_t*)alloc((size_t)ROWS_Q * D_MODEL * 2);
  float*  x2    = (float*) alloc((size_t)ROWS_Q * D_MODEL * 4);
  bf16_t* f_b   = (bf16_t*)alloc((size_t)ROWS_Q * D_MODEL * 2);
  bf16_t* h1    = (bf16_t*)alloc((size_t)ROWS_Q * DFF * 2);
  unsigned char* maskT = (unsigned char*)alloc((size_t)BATCH * LQ * LQ);

  const dim3 blk(256);
  auto cvt = [&](const float* src, bf16_t* dst, int n) {
    f32_to_bf16_kernel<<<(n + 255) / 256, blk, 0, stream>>>(src, dst, n);
  };

  // 1. memory bank passthrough -> output slabs 0..3
  (void)hipMemcpyAsync(mem_out, memb,
                       (size_t)4 * ROWS_Q * D_MODEL * sizeof(float),
                       hipMemcpyDeviceToDevice, stream);
  // 2. LN1: q (f32 -> output slab 4, bf16 -> mem_b slab 4)
  ln_kernel<<<ROWS_Q, blk, 0, stream>>>(x, ln1g, ln1b, q_f32, q_b);
  // 3. bf16 casts + mask transpose
  cvt(memb, mem_b, 4 * ROWS_Q * D_MODEL);
  cvt(Wq, Wq_b, D_MODEL * D_MODEL);
  cvt(Wk, Wk_b, D_MODEL * D_MODEL);
  cvt(Wv, Wv_b, D_MODEL * D_MODEL);
  cvt(Wo, Wo_b, D_MODEL * D_MODEL);
  cvt(W1, W1_b, D_MODEL * DFF);
  cvt(W2, W2_b, DFF * D_MODEL);
  mask_transpose_kernel<<<(BATCH * LQ * LQ) / 256, blk, 0, stream>>>(mask, maskT);
  // 4. projections (WMMA); q scale = log2(e)/sqrt(dh) -> exp2-domain softmax
  gemm_bf16_wmma<0><<<dim3(D_MODEL / 128, ROWS_Q / 64), blk, 0, stream>>>(
      q_b, Wq_b, nullptr, nullptr, 0.125f * 1.44269504f, ROWS_Q, D_MODEL,
      D_MODEL, nullptr, pqw);
  gemm_bf16_wmma<1><<<dim3(D_MODEL / 128, ROWS_MEM / 64), blk, 0, stream>>>(
      mem_b, Wk_b, nullptr, nullptr, 1.0f, ROWS_MEM, D_MODEL, D_MODEL, nullptr, pkw);
  gemm_bf16_wmma<5><<<dim3(D_MODEL / 128, ROWS_MEM / 64), blk, 0, stream>>>(
      mem_b, Wv_b, nullptr, nullptr, 1.0f, ROWS_MEM, D_MODEL, D_MODEL, nullptr, pvtw);
  // 5. flash attention: 256 blocks = (b, h, 128-row q chunk)
  attn_kernel<<<dim3(BATCH * NHEAD * (LQ / 128)), blk, 0, stream>>>(
      pqw, pkw, pvtw, maskT, ctxw);
  // 6. output projection + residual -> x2
  gemm_bf16_wmma<2><<<dim3(D_MODEL / 128, ROWS_Q / 64), blk, 0, stream>>>(
      ctxw, Wo_b, nullptr, x, 1.0f, ROWS_Q, D_MODEL, D_MODEL, x2, nullptr);
  // 7. LN2 -> f (bf16)
  ln_kernel<<<ROWS_Q, blk, 0, stream>>>(x2, ln2g, ln2b, nullptr, f_b);
  // 8. FFN1: relu(f @ W1 + b1)
  gemm_bf16_wmma<3><<<dim3(DFF / 128, ROWS_Q / 64), blk, 0, stream>>>(
      f_b, W1_b, b1, nullptr, 1.0f, ROWS_Q, DFF, D_MODEL, nullptr, h1);
  // 9. FFN2: h1 @ W2 + b2 + x2 -> out
  gemm_bf16_wmma<4><<<dim3(D_MODEL / 128, ROWS_Q / 64), blk, 0, stream>>>(
      h1, W2_b, b2, x2, 1.0f, ROWS_Q, D_MODEL, DFF, outp, nullptr);
}
